// AttentionBlock_6167573037443
// MI455X (gfx1250) — compile-verified, matrix operand order checked by reading, not run on hardware
//
#include <hip/hip_runtime.h>
#include <hip/hip_bf16.h>
#include <math.h>

#define B_   2
#define C_   1024
#define T_   2048
#define H_   16
#define CH_  64
#define G_   32
#define RP_SCALE 8.0f                       // sqrt(64)
#define QK_SCALE 0.35355339059327379f       // 64^-0.25

typedef __attribute__((ext_vector_type(16))) __bf16 bf16x16;
typedef __attribute__((ext_vector_type(8)))  __bf16 bf16x8;
typedef __attribute__((ext_vector_type(8)))  float  f32x8;
typedef __attribute__((ext_vector_type(4)))  float  f32x4;

static __device__ inline bf16x16 join8(bf16x8 lo, bf16x8 hi) {
  return __builtin_shufflevector(lo, hi, 0,1,2,3,4,5,6,7,8,9,10,11,12,13,14,15);
}
// One lane's slice of a 16x32 bf16 WMMA fragment: two contiguous 16B chunks
// (K 0..7 / 16..23 for lanes 0-15, K 8..15 / 24..31 for lanes 16-31).
static __device__ inline bf16x16 ldfrag(const __bf16* p0, const __bf16* p1) {
  return join8(*(const bf16x8*)p0, *(const bf16x8*)p1);
}
static __device__ inline f32x8 wmma_bf16(bf16x16 a, bf16x16 b, f32x8 c) {
  return __builtin_amdgcn_wmma_f32_16x16x32_bf16(false, a, false, b, (short)0, c, false, false);
}

// ---------------- GroupNorm -> bf16 xnT[B,T,C] (LDS-transposed stores) ----------------
__global__ void k_groupnorm(const float* __restrict__ x, const float* __restrict__ gw,
                            const float* __restrict__ gb, __bf16* __restrict__ xnT) {
  const int CPG = C_ / G_;           // 32
  const int N   = CPG * T_;          // 65536
  int b = blockIdx.x / G_;
  int g = blockIdx.x % G_;
  const float* base = x + ((size_t)b * C_ + (size_t)g * CPG) * T_;
  float s = 0.f, sq = 0.f;
  for (int i = threadIdx.x; i < N; i += 256) {
    float v = base[i];
    s += v; sq += v * v;
  }
  __shared__ float rs[256], rq[256];
  rs[threadIdx.x] = s; rq[threadIdx.x] = sq;
  __syncthreads();
  for (int off = 128; off > 0; off >>= 1) {
    if (threadIdx.x < off) { rs[threadIdx.x] += rs[threadIdx.x + off];
                             rq[threadIdx.x] += rq[threadIdx.x + off]; }
    __syncthreads();
  }
  float mean = rs[0] / (float)N;
  float var  = rq[0] / (float)N - mean * mean;
  float rstd = rsqrtf(var + 1e-5f);

  __shared__ __bf16 tr[64][33];      // [t-local][c-local], odd stride vs banks
  int tt = threadIdx.x & 63, cl0 = threadIdx.x >> 6;
  int wt = threadIdx.x >> 2, qd = threadIdx.x & 3;
  for (int t0 = 0; t0 < T_; t0 += 64) {
    for (int c = cl0; c < CPG; c += 4) {
      float v = (base[(size_t)c * T_ + t0 + tt] - mean) * rstd;
      v = v * gw[g * CPG + c] + gb[g * CPG + c];
      tr[tt][c] = (__bf16)v;
    }
    __syncthreads();
    bf16x8 pk;
#pragma unroll
    for (int i = 0; i < 8; i++) pk[i] = tr[wt][qd * 8 + i];
    *(bf16x8*)(xnT + ((size_t)b * T_ + t0 + wt) * C_ + g * CPG + qd * 8) = pk;
    __syncthreads();
  }
}

// ---------------- fp32 -> bf16 weight convert ----------------
__global__ void k_cvt_bf16(const float* __restrict__ in, __bf16* __restrict__ out, int n) {
  for (int i = blockIdx.x * 256 + threadIdx.x; i < n; i += gridDim.x * 256)
    out[i] = (__bf16)in[i];
}

// ---------------- relative-position bias table [H][4096] ----------------
__global__ void k_bias_table(const float* __restrict__ rel_emb, float* __restrict__ biasd) {
  int h = blockIdx.x;
  for (int d = threadIdx.x; d < 2 * T_ - 1; d += 256) {
    int rel = d - (T_ - 1);            // rel = s - t
    int n = -rel;
    int ret = (n < 0) ? 16 : 0;
    int an = (n < 0) ? -n : n;
    int bucket;
    if (an < 8) {
      bucket = ret + an;
    } else {
      int v = 8 + (int)(logf((float)an * 0.125f) * (8.0f / logf(8.0f)));
      if (v > 15) v = 15;
      bucket = ret + v;
    }
    biasd[h * 4096 + d] = rel_emb[bucket * H_ + h] * RP_SCALE;
  }
}

// ---------------- QKV GEMM: block tile 128(M) x 64(N), LDS-shared B panel ----------------
__global__ void __launch_bounds__(256) k_qkv_gemm(
    const __bf16* __restrict__ W, const float* __restrict__ bias,
    const __bf16* __restrict__ xnT,
    __bf16* __restrict__ qbuf, __bf16* __restrict__ kbuf, __bf16* __restrict__ vbuf) {
  __shared__ __align__(16) __bf16 Bsh[2][64][40];     // double-buffered B panel
  __shared__ __align__(16) __bf16 tsh[8][16][72];     // per-wave store-transpose tile
  int wave = threadIdx.x >> 5, lane = threadIdx.x & 31;
  int hl = lane >> 4, nl = lane & 15;
  int blk = blockIdx.x;
  int b    = blk / (24 * 32);
  int r2   = blk % (24 * 32);
  int o0   = (r2 / 32) * 128 + wave * 16;             // wave's 16 output rows
  int n0   = (r2 % 32) * 64;                          // block's 64 output cols

  const __bf16* Abase = W + (size_t)(o0 + nl) * C_;
  const __bf16* Bbase = xnT + (size_t)b * T_ * C_;
  int frow = threadIdx.x >> 2, fqd = threadIdx.x & 3; // B-panel fill mapping

  auto fillB = [&](int buf, int k0n) {
    *(bf16x8*)&Bsh[buf][frow][fqd * 8] =
        *(const bf16x8*)(Bbase + (size_t)(n0 + frow) * C_ + k0n + fqd * 8);
  };

  f32x8 acc[4] = {};
  fillB(0, 0);
  __syncthreads();
  for (int k0 = 0; k0 < C_; k0 += 32) {
    int cur = (k0 >> 5) & 1;
    if (k0 + 32 < C_) fillB(cur ^ 1, k0 + 32);
    __builtin_prefetch(Abase + k0 + 256, 0, 3);
    bf16x16 af = ldfrag(Abase + k0 + hl * 8, Abase + k0 + 16 + hl * 8);
#pragma unroll
    for (int j = 0; j < 4; j++) {
      const __bf16* brow = &Bsh[cur][j * 16 + nl][0];
      acc[j] = wmma_bf16(af, ldfrag(brow + hl * 8, brow + 16 + hl * 8), acc[j]);
    }
    __syncthreads();
  }

  int h    = o0 / (3 * CH_);
  int sect = o0 % (3 * CH_);
  int type = sect / CH_;        // 0=q 1=k 2=v (16-row tile never straddles)
  int chb  = sect % CH_;
  if (type < 2) {
    __bf16* dst = (type == 0) ? qbuf : kbuf;
#pragma unroll
    for (int j = 0; j < 4; j++) {
      int t = n0 + j * 16 + nl;
      bf16x8 pk;
#pragma unroll
      for (int r = 0; r < 8; r++)
        pk[r] = (__bf16)((acc[j][r] + bias[o0 + hl * 8 + r]) * QK_SCALE);
      *(bf16x8*)(dst + (((size_t)b * H_ + h) * T_ + t) * CH_ + chb + hl * 8) = pk;
    }
  } else {
    // stage [ch][t] tile in LDS, then contiguous 64B row stores into v[B,H,CH,T]
#pragma unroll
    for (int j = 0; j < 4; j++)
#pragma unroll
      for (int r = 0; r < 8; r++)
        tsh[wave][hl * 8 + r][j * 16 + nl] =
            (__bf16)(acc[j][r] + bias[o0 + hl * 8 + r]);
    __builtin_amdgcn_wave_barrier();
    int row = lane >> 1, sg = lane & 1;
    const __bf16* srow = &tsh[wave][row][sg * 32];
    __bf16* dst = vbuf + (((size_t)b * H_ + h) * CH_ + chb + row) * T_ + n0 + sg * 32;
#pragma unroll
    for (int c = 0; c < 4; c++)
      *(bf16x8*)(dst + c * 8) = *(const bf16x8*)(srow + c * 8);
  }
}

// ---------------- flash attention: block = (b,h,128 query rows), shared K/V panels ----------------
__global__ void __launch_bounds__(256) k_attention(
    const __bf16* __restrict__ qbuf, const __bf16* __restrict__ kbuf,
    const __bf16* __restrict__ vbuf, const float* __restrict__ biasd,
    __bf16* __restrict__ aT) {
  __shared__ __align__(16) __bf16 Ksh[2][32][72];     // [s-local][ch]
  __shared__ __align__(16) __bf16 Vsh[2][64][40];     // [ch][s-local]
  __shared__ __align__(16) __bf16 pbuf[8][16][40];    // per-wave P tile [t][s]
  __shared__ __align__(16) __bf16 osh[8][16][72];     // per-wave O tile  [t][ch]
  int wave = threadIdx.x >> 5, lane = threadIdx.x & 31;
  int hl = lane >> 4, nl = lane & 15;
  int blk = blockIdx.x;
  int b    = blk / (H_ * 16);
  int r2   = blk % (H_ * 16);
  int h    = r2 / 16;
  int t0   = (r2 % 16) * 128 + wave * 16;             // wave's 16 query rows

  const __bf16* qb = qbuf + ((size_t)b * H_ + h) * T_ * CH_;
  const __bf16* kb = kbuf + ((size_t)b * H_ + h) * T_ * CH_;
  const __bf16* vb = vbuf + ((size_t)b * H_ + h) * CH_ * T_;
  const float*  bt = biasd + h * 4096;

  int krow = threadIdx.x >> 3, kqd = threadIdx.x & 7; // K fill: 32 rows x 8 chunks
  int vrow = threadIdx.x >> 2, vqd = threadIdx.x & 3; // V fill: 64 rows x 4 chunks
  auto fillKV = [&](int buf, int s0n) {
    *(bf16x8*)&Ksh[buf][krow][kqd * 8] =
        *(const bf16x8*)(kb + (size_t)(s0n + krow) * CH_ + kqd * 8);
    *(bf16x8*)&Vsh[buf][vrow][vqd * 8] =
        *(const bf16x8*)(vb + (size_t)vrow * T_ + s0n + vqd * 8);
  };

  const __bf16* qrow = qb + (size_t)(t0 + nl) * CH_;
  bf16x16 qa0 = ldfrag(qrow +      hl * 8, qrow + 16 + hl * 8);   // K = ch 0..31
  bf16x16 qa1 = ldfrag(qrow + 32 + hl * 8, qrow + 48 + hl * 8);   // K = ch 32..63

  f32x8 oc[4] = {};
  float mrow[8], lrow[8];
#pragma unroll
  for (int r = 0; r < 8; r++) { mrow[r] = -INFINITY; lrow[r] = 0.f; }

  fillKV(0, 0);
  __syncthreads();
  for (int s0 = 0; s0 < T_; s0 += 32) {
    int cur = (s0 >> 5) & 1;
    if (s0 + 32 < T_) fillKV(cur ^ 1, s0 + 32);
    f32x8 sc[2];
#pragma unroll
    for (int j = 0; j < 2; j++) {
      const __bf16* kr = &Ksh[cur][j * 16 + nl][0];
      f32x8 c = {};
      c = wmma_bf16(qa0, ldfrag(kr +      hl * 8, kr + 16 + hl * 8), c);
      c = wmma_bf16(qa1, ldfrag(kr + 32 + hl * 8, kr + 48 + hl * 8), c);
      sc[j] = c;
    }
    // bias + online softmax; row m = r + hl*8 lives in this lane's half-wave
#pragma unroll
    for (int r = 0; r < 8; r++) {
      int trow = t0 + r + hl * 8;
      float v0 = sc[0][r] + bt[s0 + nl      - trow + (T_ - 1)];
      float v1 = sc[1][r] + bt[s0 + 16 + nl - trow + (T_ - 1)];
      float mx = fmaxf(v0, v1);
#pragma unroll
      for (int off = 1; off < 16; off <<= 1) mx = fmaxf(mx, __shfl_xor(mx, off, 32));
      float mnew = fmaxf(mrow[r], mx);
      float sfr  = expf(mrow[r] - mnew);
      float p0 = expf(v0 - mnew);
      float p1 = expf(v1 - mnew);
      float rsum = p0 + p1;
#pragma unroll
      for (int off = 1; off < 16; off <<= 1) rsum += __shfl_xor(rsum, off, 32);
      lrow[r] = lrow[r] * sfr + rsum;
      mrow[r] = mnew;
#pragma unroll
      for (int j = 0; j < 4; j++) oc[j][r] *= sfr;
      int row = r + hl * 8;
      pbuf[wave][row][nl]      = (__bf16)p0;          // C-layout -> LDS
      pbuf[wave][row][16 + nl] = (__bf16)p1;
    }
    __builtin_amdgcn_wave_barrier();                  // per-wave LDS ops are in-order
    const __bf16* prow = &pbuf[wave][nl][0];          // reload as A-fragment
    bf16x16 pa = ldfrag(prow + hl * 8, prow + 16 + hl * 8);
    __builtin_amdgcn_wave_barrier();
#pragma unroll
    for (int j = 0; j < 4; j++) {                     // O[t, ch-tile j] += P x V
      const __bf16* vr = &Vsh[cur][j * 16 + nl][0];
      oc[j] = wmma_bf16(pa, ldfrag(vr + hl * 8, vr + 16 + hl * 8), oc[j]);
    }
    __syncthreads();
  }
  // normalize, stage [t][ch] tile in LDS, write contiguous 64B runs to aT[B,T,C]
#pragma unroll
  for (int j = 0; j < 4; j++)
#pragma unroll
    for (int r = 0; r < 8; r++)
      osh[wave][r + hl * 8][j * 16 + nl] = (__bf16)(oc[j][r] / lrow[r]);
  __builtin_amdgcn_wave_barrier();
  int row = lane >> 1, sg = lane & 1;
  const __bf16* srow = &osh[wave][row][sg * 32];
  __bf16* dst = aT + ((size_t)b * T_ + t0 + row) * C_ + h * CH_ + sg * 32;
#pragma unroll
  for (int c = 0; c < 4; c++)
    *(bf16x8*)(dst + c * 8) = *(const bf16x8*)(srow + c * 8);
}

// ---------------- proj GEMM + residual -> fp32 out[B,C,T] ----------------
__global__ void __launch_bounds__(256) k_proj(
    const __bf16* __restrict__ W, const float* __restrict__ bias,
    const __bf16* __restrict__ aT, const float* __restrict__ x,
    float* __restrict__ out) {
  __shared__ __align__(16) __bf16 Bsh[2][64][40];     // double-buffered B panel
  __shared__ __align__(16) float  fsh[8][16][68];     // per-wave [o][t] tile
  int wave = threadIdx.x >> 5, lane = threadIdx.x & 31;
  int hl = lane >> 4, nl = lane & 15;
  int blk = blockIdx.x;
  int b    = blk / (8 * 32);
  int r2   = blk % (8 * 32);
  int o0   = (r2 / 32) * 128 + wave * 16;
  int n0   = (r2 % 32) * 64;
  const __bf16* Abase = W + (size_t)(o0 + nl) * C_;
  const __bf16* Bbase = aT + (size_t)b * T_ * C_;
  int frow = threadIdx.x >> 2, fqd = threadIdx.x & 3;

  auto fillB = [&](int buf, int k0n) {
    *(bf16x8*)&Bsh[buf][frow][fqd * 8] =
        *(const bf16x8*)(Bbase + (size_t)(n0 + frow) * C_ + k0n + fqd * 8);
  };

  f32x8 acc[4] = {};
  fillB(0, 0);
  __syncthreads();
  for (int k0 = 0; k0 < C_; k0 += 32) {
    int cur = (k0 >> 5) & 1;
    if (k0 + 32 < C_) fillB(cur ^ 1, k0 + 32);
    __builtin_prefetch(Abase + k0 + 256, 0, 3);
    bf16x16 af = ldfrag(Abase + k0 + hl * 8, Abase + k0 + 16 + hl * 8);
#pragma unroll
    for (int j = 0; j < 4; j++) {
      const __bf16* brow = &Bsh[cur][j * 16 + nl][0];
      acc[j] = wmma_bf16(af, ldfrag(brow + hl * 8, brow + 16 + hl * 8), acc[j]);
    }
    __syncthreads();
  }
  // stage fp32 tile per wave, then coalesced residual-add + store
#pragma unroll
  for (int j = 0; j < 4; j++)
#pragma unroll
    for (int r = 0; r < 8; r++)
      fsh[wave][hl * 8 + r][j * 16 + nl] = acc[j][r] + bias[o0 + hl * 8 + r];
  __builtin_amdgcn_wave_barrier();
  int row = lane >> 1, sg = lane & 1;
  size_t idx = ((size_t)b * C_ + o0 + row) * T_ + n0 + sg * 32;
#pragma unroll
  for (int c = 0; c < 8; c++) {
    f32x4 xa = *(const f32x4*)(x + idx + c * 4);
    f32x4 va = *(const f32x4*)&fsh[wave][row][sg * 32 + c * 4];
    *(f32x4*)(out + idx + c * 4) = xa + va;
  }
}

extern "C" void kernel_launch(void* const* d_in, const int* in_sizes, int n_in,
                              void* d_out, int out_size, void* d_ws, size_t ws_size,
                              hipStream_t stream) {
  const float* x       = (const float*)d_in[0];
  const float* gn_w    = (const float*)d_in[1];
  const float* gn_b    = (const float*)d_in[2];
  const float* qkv_w   = (const float*)d_in[3];
  const float* qkv_b   = (const float*)d_in[4];
  const float* proj_w  = (const float*)d_in[5];
  const float* proj_b  = (const float*)d_in[6];
  const float* rel_emb = (const float*)d_in[7];
  float* out = (float*)d_out;

  char* ws = (char*)d_ws;
  size_t off = 0;
  auto alloc = [&](size_t bytes) -> void* {
    void* p = ws + off;
    off += (bytes + 255) & ~(size_t)255;
    return p;
  };
  __bf16* xnT   = (__bf16*)alloc((size_t)B_ * T_ * C_ * 2);        // 8.39 MB
  __bf16* qb    = (__bf16*)alloc((size_t)B_ * H_ * T_ * CH_ * 2);  // 8.39 MB
  __bf16* kb    = (__bf16*)alloc((size_t)B_ * H_ * T_ * CH_ * 2);  // 8.39 MB
  __bf16* vb    = (__bf16*)alloc((size_t)B_ * H_ * CH_ * T_ * 2);  // 8.39 MB
  __bf16* aT    = (__bf16*)alloc((size_t)B_ * T_ * C_ * 2);        // 8.39 MB
  __bf16* wqkv  = (__bf16*)alloc((size_t)3 * C_ * C_ * 2);         // 6.29 MB
  __bf16* wproj = (__bf16*)alloc((size_t)C_ * C_ * 2);             // 2.10 MB
  float*  biasd = (float*)alloc((size_t)H_ * 4096 * 4);            // 0.26 MB
  (void)ws_size; (void)in_sizes; (void)n_in; (void)out_size;

  k_groupnorm<<<B_ * G_, 256, 0, stream>>>(x, gn_w, gn_b, xnT);
  k_cvt_bf16<<<1024, 256, 0, stream>>>(qkv_w, wqkv, 3 * C_ * C_);
  k_cvt_bf16<<<512, 256, 0, stream>>>(proj_w, wproj, C_ * C_);
  k_bias_table<<<H_, 256, 0, stream>>>(rel_emb, biasd);

  // block tile 128x64, 24*32 tiles/batch -> 1536 blocks
  k_qkv_gemm<<<1536, 256, 0, stream>>>(wqkv, qkv_b, xnT, qb, kb, vb);
  // (b,h,128-query-row) blocks -> 2*16*16 = 512 blocks
  k_attention<<<512, 256, 0, stream>>>(qb, kb, vb, biasd, aT);
  // block tile 128x64, 8*32 tiles/batch -> 512 blocks
  k_proj<<<512, 256, 0, stream>>>(wproj, proj_b, aT, x, out);
}
